// UltraThermalLSTM_5291399708818
// MI455X (gfx1250) — matrix operand order repaired and checked by reading, not compiled, not run on hardware
//
#include <hip/hip_runtime.h>
#include <hip/hip_bf16.h>
#include <stdint.h>

// ---------------------------------------------------------------------------
// Model dims (UltraThermalLSTM)
// ---------------------------------------------------------------------------
#define B_   256
#define L_   512
#define D_   64
#define P_   256
#define H_   512
#define M_   256
#define HOR_ 9
#define J_   12
#define G4H  (4 * H_)   // 2048

typedef __attribute__((ext_vector_type(16))) _Float16 v16h;
typedef __attribute__((ext_vector_type(8)))  float    v8f;

// int4 vector + address-space-qualified flavors for the async LDS builtin:
// prototype (from clang diagnostic): (as1 int4* src, as3 int4* dst, Ii, Ii)
typedef int v4i_ __attribute__((vector_size(4 * sizeof(int))));
typedef __attribute__((address_space(1))) v4i_ as1_v4i;
typedef __attribute__((address_space(3))) v4i_ as3_v4i;

#if defined(__gfx1250__) &&                                          \
    __has_builtin(__builtin_amdgcn_global_load_async_to_lds_b128) && \
    __has_builtin(__builtin_amdgcn_s_wait_asynccnt)
#define USE_ASYNC_LDS 1
#else
#define USE_ASYNC_LDS 0
#endif

#if USE_ASYNC_LDS
__device__ __forceinline__ void async_copy_b128(const _Float16* gsrc,
                                                _Float16* ldst) {
  __builtin_amdgcn_global_load_async_to_lds_b128(
      (as1_v4i*)gsrc, (as3_v4i*)ldst, 0, 0);
}
#endif

// ---------------------------------------------------------------------------
// WMMA fragment packing helpers (CDNA5 16-bit layouts, wave32)
//  A (16x32, MxK): lane m = lane&15, half = lane>>4.
//    j in [0,8):  k = 8*half + j          (contiguous 8 halves)
//    j in [8,16): k = 16 + 8*half + (j-8) (contiguous 8 halves, +16)
//  B (32x16, KxN): lane n = lane&15, half = lane>>4.
//    j in [0,8):  k = 8*half + j; j in [8,16): k = 16 + 8*half + (j-8)
//  C/D (16x16 f32): lane n = lane&15; vgpr v: m = v + 8*(lane>>4)
// ---------------------------------------------------------------------------
__device__ __forceinline__ v16h frag_a16(const _Float16* X, int ld,
                                         int m, int half, int k0) {
  const _Float16* p0 = X + (size_t)m * ld + k0 + half * 8;
  const _Float16* p1 = p0 + 16;
  v16h a;
#pragma unroll
  for (int j = 0; j < 8; ++j) { a[j] = p0[j]; a[8 + j] = p1[j]; }
  return a;
}

__device__ __forceinline__ v16h frag_b16(const _Float16* __restrict__ W,
                                         int ld, int n, int half, int k0) {
  const _Float16* p = W + (size_t)n * ld + k0 + half * 8;
  v16h b;
#pragma unroll
  for (int j = 0; j < 8; ++j) { b[j] = p[j]; b[8 + j] = p[16 + j]; }
  return b;
}

__device__ __forceinline__ float sigmoidf_(float x) {
  return 1.0f / (1.0f + expf(-x));
}
__device__ __forceinline__ float gelu_exact(float x) {
  return 0.5f * x * (1.0f + erff(x * 0.70710678118654752440f));
}

// ---------------------------------------------------------------------------
// Small utility kernels
// ---------------------------------------------------------------------------
__global__ void cvt_f32_to_f16_kernel(const float* __restrict__ src,
                                      _Float16* __restrict__ dst, int n) {
  int i = blockIdx.x * blockDim.x + threadIdx.x;
  if (i < n) dst[i] = (_Float16)src[i];
}

// Wp (64,256) f32 -> WpT (256,64) f16  (so proj B-fragments are contiguous)
__global__ void cvt_transpose_wp_kernel(const float* __restrict__ Wp,
                                        _Float16* __restrict__ WpT) {
  int i = blockIdx.x * blockDim.x + threadIdx.x;
  if (i < D_ * P_) {
    int k = i / P_;
    int p = i % P_;
    WpT[(size_t)p * D_ + k] = (_Float16)Wp[i];
  }
}

__global__ void bias_sum_kernel(const float* __restrict__ a,
                                const float* __restrict__ b,
                                float* __restrict__ out, int n) {
  int i = blockIdx.x * blockDim.x + threadIdx.x;
  if (i < n) out[i] = a[i] + b[i];
}

__global__ void zero_u32_kernel(uint32_t* __restrict__ p, int nwords) {
  int i = blockIdx.x * blockDim.x + threadIdx.x;
  if (i < nwords) p[i] = 0u;
}

// ---------------------------------------------------------------------------
// Kernel 1: h = gelu(LN(x @ Wp + bp)) -> f16   (B*L rows of 256)
// 512 threads = 16 waves; block computes a 16-row x 256-col tile.
// ---------------------------------------------------------------------------
__global__ __launch_bounds__(512) void proj_ln_gelu_kernel(
    const float* __restrict__ x,        // (B*L, 64)
    const _Float16* __restrict__ WpT,   // (256, 64) f16 (transposed)
    const float* __restrict__ bp,       // (256,)
    const float* __restrict__ gamma,    // (256,)
    const float* __restrict__ beta,     // (256,)
    _Float16* __restrict__ h0)          // (B*L, 256)
{
  __shared__ float tile[16][P_];  // 16 KB

  const int tid  = threadIdx.x;
  const int wave = tid >> 5;
  const int lane = tid & 31;
  const int n15  = lane & 15;
  const int half = lane >> 4;
  const int r0   = blockIdx.x * 16;   // first global row
  const int n0   = wave * 16;         // first col of this wave's tile

  v8f acc = {};
#pragma unroll
  for (int k0 = 0; k0 < D_; k0 += 32) {
    // A fragment from f32 x, converting to f16 on the fly
    v16h a;
    const float* pa = x + (size_t)(r0 + n15) * D_ + k0;
#pragma unroll
    for (int j = 0; j < 8; ++j) {
      a[j]     = (_Float16)pa[8 * half + j];
      a[8 + j] = (_Float16)pa[16 + 8 * half + j];
    }
    // B fragment from pre-transposed f16 WpT: two runs of 8 halves
    v16h b = frag_b16(WpT, D_, n0 + n15, half, k0);

    acc = __builtin_amdgcn_wmma_f32_16x16x32_f16(
        false, a, false, b, (short)0, acc, false, false);
  }

  const float bpv = bp[n0 + n15];
#pragma unroll
  for (int v = 0; v < 8; ++v) {
    int m = v + 8 * half;
    tile[m][n0 + n15] = acc[v] + bpv;
  }
  __syncthreads();

  // LayerNorm + GELU: wave w handles row w (32 lanes x 8 cols each)
  const int m = wave;
  float vals[8];
  float s = 0.0f;
#pragma unroll
  for (int u = 0; u < 8; ++u) { vals[u] = tile[m][lane * 8 + u]; s += vals[u]; }
#pragma unroll
  for (int off = 16; off >= 1; off >>= 1) s += __shfl_xor(s, off, 32);
  const float mu = s * (1.0f / 256.0f);
  float vsum = 0.0f;
#pragma unroll
  for (int u = 0; u < 8; ++u) { float d = vals[u] - mu; vsum += d * d; }
#pragma unroll
  for (int off = 16; off >= 1; off >>= 1) vsum += __shfl_xor(vsum, off, 32);
  const float rstd = rsqrtf(vsum * (1.0f / 256.0f) + 1e-5f);

#pragma unroll
  for (int u = 0; u < 8; ++u) {
    int col  = lane * 8 + u;
    float hv = gamma[col] * (vals[u] - mu) * rstd + beta[col];
    h0[(size_t)(r0 + m) * P_ + col] = (_Float16)gelu_exact(hv);
  }
}

// ---------------------------------------------------------------------------
// Kernel 2: one fused LSTM timestep for one layer (KX = input width).
//   gates = Xin_t @ Wih^T + Hprev @ Whh^T + bias   (B x 4H)
//   c = sigmoid(f)*c + sigmoid(i)*tanh(g);  h = sigmoid(o)*tanh(c)
//
// Grid: (B/16) * (H/64) = 128 blocks; 256 threads = 8 waves.
// Per block: 16 batch rows x 64 hidden cols. Wave w: quadrant q = w&3,
// col-pair s = w>>2; two 16x16 WMMA tiles per A-fragment.
// Activation strips (X_t, H_prev) staged into LDS once per step via
// GLOBAL_LOAD_ASYNC_TO_LDS (ASYNCcnt) when available.
// ---------------------------------------------------------------------------
template <int KX>
__global__ __launch_bounds__(256) void lstm_step_kernel(
    const _Float16* __restrict__ Xin,   // rows stride ldx, KX cols used
    int ldx,
    const _Float16* __restrict__ Hprev, // (B, 512)
    const _Float16* __restrict__ Wih,   // (4H, KX) f16
    const _Float16* __restrict__ Whh,   // (4H, 512) f16
    const float*    __restrict__ bias,  // (4H,) = bih + bhh
    float*          __restrict__ Cst,   // (B, 512) f32 cell state (in/out)
    _Float16*       __restrict__ Hnew)  // (B, 512) f16 hidden out
{
  __shared__ _Float16 xs[16 * KX];        // X_t strip  (8/16 KB)
  __shared__ _Float16 hsh[16 * H_];       // H_prev strip (16 KB)
  __shared__ float    gsh[4][16][64];     // gate tiles (16 KB)

  const int tid  = threadIdx.x;
  const int wave = tid >> 5;
  const int lane = tid & 31;
  const int n15  = lane & 15;
  const int half = lane >> 4;

  const int rb0  = (blockIdx.x >> 3) * 16;  // batch-row tile (16 tiles)
  const int cg   = blockIdx.x & 7;          // 64-wide hidden-col group
  const int quad = wave & 3;                // gate: 0=i 1=f 2=g 3=o
  const int sub  = wave >> 2;               // 32-col pair within group

  // gate columns of this wave's two tiles
  const int g0 = quad * H_ + cg * 64 + sub * 32 + n15;
  const int g1 = g0 + 16;

  // ---- stage activation strips into LDS (16B chunks per lane) ----
#if USE_ASYNC_LDS
  for (int i = tid; i < (16 * KX) / 8; i += 256) {
    int r  = i / (KX / 8);
    int c8 = i % (KX / 8);
    async_copy_b128(Xin + (size_t)(rb0 + r) * ldx + (size_t)c8 * 8,
                    xs + (size_t)i * 8);
  }
  for (int i = tid; i < (16 * H_) / 8; i += 256) {
    int r  = i / (H_ / 8);
    int c8 = i % (H_ / 8);
    async_copy_b128(Hprev + (size_t)(rb0 + r) * H_ + (size_t)c8 * 8,
                    hsh + (size_t)i * 8);
  }
  __builtin_amdgcn_s_wait_asynccnt(0);
#else
  for (int i = tid; i < (16 * KX) / 8; i += 256) {
    int r  = i / (KX / 8);
    int c8 = i % (KX / 8);
    ((uint4*)xs)[i] =
        *(const uint4*)(Xin + (size_t)(rb0 + r) * ldx + (size_t)c8 * 8);
  }
  for (int i = tid; i < (16 * H_) / 8; i += 256) {
    int r  = i / (H_ / 8);
    int c8 = i % (H_ / 8);
    ((uint4*)hsh)[i] =
        *(const uint4*)(Hprev + (size_t)(rb0 + r) * H_ + (size_t)c8 * 8);
  }
#endif
  __syncthreads();

  v8f acc0 = {};
  v8f acc1 = {};

  // Input-to-hidden contribution (A from LDS, B streamed from L2)
#pragma unroll 4
  for (int k0 = 0; k0 < KX; k0 += 32) {
    v16h a  = frag_a16(xs, KX, n15, half, k0);
    v16h b0 = frag_b16(Wih, KX, g0, half, k0);
    v16h b1 = frag_b16(Wih, KX, g1, half, k0);
    acc0 = __builtin_amdgcn_wmma_f32_16x16x32_f16(
        false, a, false, b0, (short)0, acc0, false, false);
    acc1 = __builtin_amdgcn_wmma_f32_16x16x32_f16(
        false, a, false, b1, (short)0, acc1, false, false);
  }
  // Hidden-to-hidden contribution
#pragma unroll 4
  for (int k0 = 0; k0 < H_; k0 += 32) {
    v16h a  = frag_a16(hsh, H_, n15, half, k0);
    v16h b0 = frag_b16(Whh, H_, g0, half, k0);
    v16h b1 = frag_b16(Whh, H_, g1, half, k0);
    acc0 = __builtin_amdgcn_wmma_f32_16x16x32_f16(
        false, a, false, b0, (short)0, acc0, false, false);
    acc1 = __builtin_amdgcn_wmma_f32_16x16x32_f16(
        false, a, false, b1, (short)0, acc1, false, false);
  }

  const float bv0 = bias[g0];
  const float bv1 = bias[g1];
#pragma unroll
  for (int v = 0; v < 8; ++v) {
    int m = v + 8 * half;
    gsh[quad][m][sub * 32 + n15]      = acc0[v] + bv0;
    gsh[quad][m][sub * 32 + 16 + n15] = acc1[v] + bv1;
  }
  __syncthreads();

  // Cell update: 16 rows x 64 cols = 1024 elements over 256 threads
  for (int e = tid; e < 16 * 64; e += 256) {
    int r  = e >> 6;
    int nl = e & 63;
    float iv = sigmoidf_(gsh[0][r][nl]);
    float fv = sigmoidf_(gsh[1][r][nl]);
    float gv = tanhf(gsh[2][r][nl]);
    float ov = sigmoidf_(gsh[3][r][nl]);
    size_t idx = (size_t)(rb0 + r) * H_ + cg * 64 + nl;
    float c = fv * Cst[idx] + iv * gv;
    Cst[idx] = c;
    Hnew[idx] = (_Float16)(ov * tanhf(c));
  }
}

// ---------------------------------------------------------------------------
// Kernel 3: joint head. One block per batch row b (j varies per row).
//   z = gelu(h_last[b] @ Wh1[j] + bh1[j]);  out[b] = z @ Wh2[j] + bh2[j]
// ---------------------------------------------------------------------------
__global__ __launch_bounds__(256) void head_kernel(
    const _Float16* __restrict__ hlast,  // (B, 512)
    const int*      __restrict__ jidx,   // (B,)
    const float*    __restrict__ Wh1,    // (J, 512, 256)
    const float*    __restrict__ bh1,    // (J, 256)
    const float*    __restrict__ Wh2,    // (J, 256, 9)
    const float*    __restrict__ bh2,    // (J, 9)
    float*          __restrict__ out)    // (B, 9)
{
  __shared__ float z[M_];
  const int b = blockIdx.x;
  const int m = threadIdx.x;
  const int j = jidx[b];

  const _Float16* hb = hlast + (size_t)b * H_;
  const float*    W1 = Wh1 + (size_t)j * H_ * M_;
  float s = bh1[(size_t)j * M_ + m];
  for (int k = 0; k < H_; ++k) s += (float)hb[k] * W1[(size_t)k * M_ + m];
  z[m] = gelu_exact(s);
  __syncthreads();

  if (m < HOR_) {
    const float* W2 = Wh2 + (size_t)j * M_ * HOR_;
    float o = bh2[(size_t)j * HOR_ + m];
    for (int k = 0; k < M_; ++k) o += z[k] * W2[(size_t)k * HOR_ + m];
    out[(size_t)b * HOR_ + m] = o;
  }
}

// ---------------------------------------------------------------------------
// Host-side orchestration
// ---------------------------------------------------------------------------
extern "C" void kernel_launch(void* const* d_in, const int* in_sizes, int n_in,
                              void* d_out, int out_size, void* d_ws, size_t ws_size,
                              hipStream_t stream) {
  (void)in_sizes; (void)n_in; (void)out_size; (void)ws_size;

  const float* x     = (const float*)d_in[0];
  const int*   jidx  = (const int*)  d_in[1];
  const float* Wp    = (const float*)d_in[2];
  const float* bp    = (const float*)d_in[3];
  const float* gamma = (const float*)d_in[4];
  const float* beta  = (const float*)d_in[5];
  const float* Wih0  = (const float*)d_in[6];
  const float* Whh0  = (const float*)d_in[7];
  const float* bih0  = (const float*)d_in[8];
  const float* bhh0  = (const float*)d_in[9];
  const float* Wih1  = (const float*)d_in[10];
  const float* Whh1  = (const float*)d_in[11];
  const float* bih1  = (const float*)d_in[12];
  const float* bhh1  = (const float*)d_in[13];
  const float* Wh1   = (const float*)d_in[14];
  const float* bh1   = (const float*)d_in[15];
  const float* Wh2   = (const float*)d_in[16];
  const float* bh2   = (const float*)d_in[17];
  float* out = (float*)d_out;

  // --- workspace carve-up (256-byte aligned chunks) ---
  uint8_t* ws = (uint8_t*)d_ws;
  size_t off = 0;
  auto carve = [&](size_t bytes) -> uint8_t* {
    uint8_t* p = ws + off;
    off = (off + bytes + 255) & ~(size_t)255;
    return p;
  };
  _Float16* h0seq = (_Float16*)carve((size_t)B_ * L_ * P_ * 2);   // 64 MiB
  _Float16* wpt   = (_Float16*)carve((size_t)P_ * D_ * 2);
  _Float16* wih0h = (_Float16*)carve((size_t)G4H * P_ * 2);
  _Float16* whh0h = (_Float16*)carve((size_t)G4H * H_ * 2);
  _Float16* wih1h = (_Float16*)carve((size_t)G4H * H_ * 2);
  _Float16* whh1h = (_Float16*)carve((size_t)G4H * H_ * 2);
  float*    bias0 = (float*)   carve((size_t)G4H * 4);
  float*    bias1 = (float*)   carve((size_t)G4H * 4);
  float*    c0    = (float*)   carve((size_t)B_ * H_ * 4);
  float*    c1    = (float*)   carve((size_t)B_ * H_ * 4);
  _Float16* h0A   = (_Float16*)carve((size_t)B_ * H_ * 2);
  _Float16* h0B   = (_Float16*)carve((size_t)B_ * H_ * 2);
  _Float16* h1A   = (_Float16*)carve((size_t)B_ * H_ * 2);
  _Float16* h1B   = (_Float16*)carve((size_t)B_ * H_ * 2);

  // --- weight preconversion f32 -> f16, bias folding, state zeroing ---
  auto cvt = [&](const float* s, _Float16* d, int n) {
    cvt_f32_to_f16_kernel<<<(n + 255) / 256, 256, 0, stream>>>(s, d, n);
  };
  cvt(Wih0, wih0h, G4H * P_);
  cvt(Whh0, whh0h, G4H * H_);
  cvt(Wih1, wih1h, G4H * H_);
  cvt(Whh1, whh1h, G4H * H_);
  cvt_transpose_wp_kernel<<<(D_ * P_ + 255) / 256, 256, 0, stream>>>(Wp, wpt);
  bias_sum_kernel<<<(G4H + 255) / 256, 256, 0, stream>>>(bih0, bhh0, bias0, G4H);
  bias_sum_kernel<<<(G4H + 255) / 256, 256, 0, stream>>>(bih1, bhh1, bias1, G4H);

  auto zero = [&](void* p, size_t bytes) {
    int nw = (int)(bytes / 4);
    zero_u32_kernel<<<(nw + 255) / 256, 256, 0, stream>>>((uint32_t*)p, nw);
  };
  zero(c0, (size_t)B_ * H_ * 4);
  zero(c1, (size_t)B_ * H_ * 4);
  zero(h0A, (size_t)B_ * H_ * 2);
  zero(h0B, (size_t)B_ * H_ * 2);
  zero(h1A, (size_t)B_ * H_ * 2);
  zero(h1B, (size_t)B_ * H_ * 2);

  // --- projection + LN + GELU ---
  proj_ln_gelu_kernel<<<(B_ * L_) / 16, 512, 0, stream>>>(
      x, wpt, bp, gamma, beta, h0seq);

  // --- recurrent chain: layer0 then layer1 at each timestep ---
  const int step_grid = (B_ / 16) * (H_ / 64);  // 128 blocks
  for (int t = 0; t < L_; ++t) {
    _Float16* h0prev = (t & 1) ? h0B : h0A;
    _Float16* h0new  = (t & 1) ? h0A : h0B;
    _Float16* h1prev = (t & 1) ? h1B : h1A;
    _Float16* h1new  = (t & 1) ? h1A : h1B;

    // Layer 0: Xin = projected sequence at time t (row stride L*P)
    lstm_step_kernel<P_><<<step_grid, 256, 0, stream>>>(
        h0seq + (size_t)t * P_, L_ * P_,
        h0prev, wih0h, whh0h, bias0, c0, h0new);

    // Layer 1: Xin = layer0's fresh hidden state (row stride H)
    lstm_step_kernel<H_><<<step_grid, 256, 0, stream>>>(
        h0new, H_,
        h1prev, wih1h, whh1h, bias1, c1, h1new);
  }

  // t = L-1 = 511 is odd -> final layer1 hidden lives in h1A
  head_kernel<<<B_, 256, 0, stream>>>(h1A, jidx, Wh1, bh1, Wh2, bh2, out);
}
